// VoxelBackBone8x_7773890806392
// MI455X (gfx1250) — matrix operand order repaired and checked by reading, not compiled, not run on hardware
//
#include <hip/hip_runtime.h>
#include <math.h>

// ---------------------------------------------------------------------------
// VoxelBackBone8x for MI455X (gfx1250, wave32, WMMA).
// Fully-templated implicit-GEMM conv3d: compile-time layer configs ->
// unrolled K-loops of v_wmma_f32_16x16x32_f16, no runtime int division.
// ---------------------------------------------------------------------------

typedef __attribute__((ext_vector_type(16))) _Float16 v16h;
typedef __attribute__((ext_vector_type(8)))  float    v8f;

union Frag16 { v16h v; uint4 q[2]; };

// ---------------------------------------------------------------------------
// Weight prepack: f32 [kz][ky][kx][cin][cout] -> f16 B fragments.
// dst[(((kb*ntiles+ntile)*32+lane)*16+h)] = W[k=kb*32+(lane>>4)*16+h][n=ntile*16+(lane&15)]
// B in contiguous k order, same as the A staging: lanes 0-15 K=0..15, lanes 16-31 K=16..31.
// ---------------------------------------------------------------------------
__global__ void prepack_wmma_b(const float* __restrict__ w, _Float16* __restrict__ dst,
                               int cin, int cout, int ntaps, int kp) {
  int idx = blockIdx.x * blockDim.x + threadIdx.x;
  int ntiles = (cout + 15) >> 4;
  int total = kp * ntiles * 16;
  if (idx >= total) return;
  int h     = idx & 15;
  int lane  = (idx >> 4) & 31;
  int rest  = idx >> 9;            // kb*ntiles + ntile
  int ntile = rest % ntiles;
  int kb    = rest / ntiles;
  int k = (kb << 5) + ((lane >> 4) << 4) + h;   // flat k = tap*cin + c
  int n = (ntile << 4) + (lane & 15);
  float val = 0.0f;
  if (k < ntaps * cin && n < cout)
    val = w[(size_t)k * cout + n];              // [(tap*cin+c)*cout + n]
  dst[idx] = (_Float16)val;
}

// ---------------------------------------------------------------------------
// Scatter voxel features into dense f16 grid + set full-res mask.
// ---------------------------------------------------------------------------
__global__ void scatter_voxels(const float* __restrict__ vf, const int* __restrict__ coords,
                               _Float16* __restrict__ grid4, unsigned char* __restrict__ m1,
                               int N) {
  int i = blockIdx.x * blockDim.x + threadIdx.x;
  if (i >= N) return;
  int z = coords[i*4+1], y = coords[i*4+2], x = coords[i*4+3];
  size_t v = ((size_t)z * 160 + y) * 160 + x;
  m1[v] = 1;
#pragma unroll
  for (int c = 0; c < 4; ++c) grid4[v*4 + c] = (_Float16)vf[i*4 + c];
}

// ---------------------------------------------------------------------------
// Mask dilation (== conv of mask with ones kernel, thresholded > 0).
// Cheap (runs on small mask arrays); kept runtime-parameterized.
// ---------------------------------------------------------------------------
struct DilP {
  const unsigned char* in;
  unsigned char* out;
  int iz, iy, ix, oz, oy, ox;
  int kz, ky, kx, sz, sy, sx, pz, py, px;
};

__global__ void mask_dilate(DilP p) {
  int v = blockIdx.x * blockDim.x + threadIdx.x;
  int nv = p.oz * p.oy * p.ox;
  if (v >= nv) return;
  int oyx = p.oy * p.ox;
  int z = v / oyx, rr = v - z * oyx, y = rr / p.ox, x = rr - y * p.ox;
  unsigned char r = 0;
  for (int dz = 0; dz < p.kz && !r; ++dz)
    for (int dy = 0; dy < p.ky && !r; ++dy)
      for (int dx = 0; dx < p.kx && !r; ++dx) {
        int izc = z * p.sz - p.pz + dz;
        int iyc = y * p.sy - p.py + dy;
        int ixc = x * p.sx - p.px + dx;
        if ((unsigned)izc < (unsigned)p.iz && (unsigned)iyc < (unsigned)p.iy &&
            (unsigned)ixc < (unsigned)p.ix &&
            p.in[(izc * p.iy + iyc) * p.ix + ixc]) r = 1;
      }
  p.out[v] = r;
}

// ---------------------------------------------------------------------------
// Sparse implicit-GEMM conv3d + BN + ReLU + mask via v_wmma_f32_16x16x32_f16.
// All layer parameters are template constants: divisions fold to shifts/magic
// multiplies, the K-loop fully unrolls (NKB wmma ops back-to-back), LDS tile
// is statically sized. Block = 128 threads = 4 wave32; M-tile = 16 output
// voxels staged cooperatively in LDS; each wave owns one 16-channel N-tile.
// Empty-mask tiles exit early (outputs pre-zeroed by memset); masked-out rows
// inside a live tile stage zeros without any global reads.
// ---------------------------------------------------------------------------
template<int CIN, int COUT, int KZ, int KY, int KX,
         int SZ, int SY, int SX, int PZ, int PY, int PX,
         int IZ, int IY, int IX, int OZ, int OY, int OX, int F32OUT>
__global__ __launch_bounds__(128) void conv_wmma_t(
    const _Float16* __restrict__ in, void* __restrict__ out,
    const _Float16* __restrict__ wpack, const unsigned char* __restrict__ mask,
    const float* __restrict__ bnw, const float* __restrict__ bnb,
    const float* __restrict__ bnm, const float* __restrict__ bnv) {
  constexpr int NTAPS  = KZ * KY * KX;
  constexpr int KREAL  = NTAPS * CIN;
  constexpr int KP     = (KREAL + 31) & ~31;
  constexpr int NKB    = KP / 32;
  constexpr int NTILES = COUT / 16;
  constexpr int NVOX   = OZ * OY * OX;
  constexpr int OYX    = OY * OX;
  constexpr int KYX    = KY * KX;
  static_assert(NVOX % 16 == 0, "tile-exact grids");
  static_assert(COUT % 16 == 0, "channel-exact tiles");

  __shared__ __align__(16) _Float16 sA[16 * KP];   // im2col tile [16][KP]
  __shared__ float sMask[16];
  __shared__ int   sAny;

  const int tid  = threadIdx.x;
  const int lane = tid & 31;
  const int wave = tid >> 5;
  const int vox0 = blockIdx.x * 16;

  if (tid < 16) sMask[tid] = mask[vox0 + tid] ? 1.0f : 0.0f;
  __syncthreads();
  if (tid == 0) {
    int a = 0;
#pragma unroll
    for (int i = 0; i < 16; ++i) a |= (sMask[i] != 0.0f);
    sAny = a;
  }
  __syncthreads();
  if (!sAny) return;     // block-uniform: whole tile masked out -> stays zero

  // ---- stage im2col patch: sA[m][tap*CIN + c], tail zero-padded to KP ----
  for (int item = tid; item < 16 * NTAPS; item += 128) {
    const int m = item / NTAPS, t = item - m * NTAPS;     // const divisors
    _Float16* dst = sA + m * KP + t * CIN;
    const _Float16* src = nullptr;
    if (sMask[m] != 0.0f) {
      const int v = vox0 + m;
      const int z = v / OYX, rr = v - z * OYX, y = rr / OX, x = rr - y * OX;
      const int dz = t / KYX, r2 = t - dz * KYX, dy = r2 / KX, dx = r2 - dy * KX;
      const int izc = z * SZ - PZ + dz;
      const int iyc = y * SY - PY + dy;
      const int ixc = x * SX - PX + dx;
      if ((unsigned)izc < (unsigned)IZ && (unsigned)iyc < (unsigned)IY &&
          (unsigned)ixc < (unsigned)IX)
        src = in + (size_t)((izc * IY + iyc) * IX + ixc) * CIN;
    }
    if constexpr ((CIN & 7) == 0) {
#pragma unroll
      for (int c = 0; c < CIN; c += 8) {
        uint4 d = src ? *(const uint4*)(src + c) : make_uint4(0u, 0u, 0u, 0u);
        *(uint4*)(dst + c) = d;
      }
    } else {
#pragma unroll
      for (int c = 0; c < CIN; ++c) dst[c] = src ? src[c] : (_Float16)0.0f;
    }
  }
  if constexpr (KP > KREAL) {
    constexpr int PER = KP - KREAL;
    for (int idx2 = tid; idx2 < 16 * PER; idx2 += 128) {
      const int m = idx2 / PER, k = KREAL + (idx2 - m * PER);
      sA[m * KP + k] = (_Float16)0.0f;
    }
  }
  __syncthreads();

  // ---- GEMM: each wave computes one 16x16 (M x N) tile ----
  const int ntile = blockIdx.y * 4 + wave;
  if (ntile >= NTILES) return;   // wave-uniform -> EXEC stays all-ones in WMMA

  v8f acc = {0.f, 0.f, 0.f, 0.f, 0.f, 0.f, 0.f, 0.f};
  const int g = lane >> 4;
  const _Float16* arow = sA + (lane & 15) * KP;     // A layout: lane&15 = M
  const uint4* wq = (const uint4*)wpack + ((size_t)ntile * 32 + lane) * 2;

#pragma unroll
  for (int kb = 0; kb < NKB; ++kb) {
    Frag16 a, b;
    const int ko = (kb << 5) + (g << 4);
    a.q[0] = *(const uint4*)(arow + ko);          // halves K=ko..ko+7
    a.q[1] = *(const uint4*)(arow + ko + 8);      // halves K=ko+8..ko+15
    b.q[0] = wq[(size_t)kb * (NTILES * 64) + 0];  // 16 contiguous halves/lane
    b.q[1] = wq[(size_t)kb * (NTILES * 64) + 1];
    if ((kb & 7) == 0 && kb + 8 < NKB)            // folded during unroll
      __builtin_prefetch((const void*)(wq + (size_t)(kb + 8) * (NTILES * 64)), 0, 2);
    acc = __builtin_amdgcn_wmma_f32_16x16x32_f16(
        false, a.v, false, b.v, (short)0, acc, false, false);
  }

  // ---- epilogue: BN + ReLU + mask, store f16 (f32 for the final layer) ----
  const int n = (ntile << 4) + (lane & 15);       // C/D layout: lane&15 = N
  const float inv   = bnw[n] * rsqrtf(bnv[n] + 1.0e-3f);
  const float shift = bnb[n] - bnm[n] * inv;
#pragma unroll
  for (int r = 0; r < 8; ++r) {
    const int m = (g << 3) + r;                   // C/D: VGPR r -> M = r + 8*g
    float val = fmaxf(acc[r] * inv + shift, 0.0f) * sMask[m];
    const size_t o = (size_t)(vox0 + m) * COUT + n;
    if constexpr (F32OUT) ((float*)out)[o] = val;
    else                  ((_Float16*)out)[o] = (_Float16)val;
  }
}

// ---------------------------------------------------------------------------
// Gather point features (64ch) from full-res grid + emit point coords.
// ---------------------------------------------------------------------------
__global__ void gather_points(const _Float16* __restrict__ xp, const int* __restrict__ coords,
                              float* __restrict__ pf, float* __restrict__ pc, int N) {
  int t = blockIdx.x * blockDim.x + threadIdx.x;
  if (t >= N * 8) return;
  int i = t >> 3, c0 = (t & 7) * 8;
  int b = coords[i*4+0], z = coords[i*4+1], y = coords[i*4+2], x = coords[i*4+3];
  size_t v = ((size_t)z * 160 + y) * 160 + x;
  const _Float16* s = xp + v * 64 + c0;
#pragma unroll
  for (int c = 0; c < 8; ++c) pf[(size_t)i * 64 + c0 + c] = (float)s[c];
  if ((t & 7) == 0) {
    pc[i*4+0] = (float)b;
    pc[i*4+1] = ((float)x + 0.5f) * 0.4f + 0.0f;
    pc[i*4+2] = ((float)y + 0.5f) * 0.4f - 32.0f;
    pc[i*4+3] = ((float)z + 0.5f) * 0.2f - 3.0f;
  }
}

// ---------------------------------------------------------------------------
// Host driver
// ---------------------------------------------------------------------------
extern "C" void kernel_launch(void* const* d_in, const int* in_sizes, int n_in,
                              void* d_out, int out_size, void* d_ws, size_t ws_size,
                              hipStream_t stream) {
  // Input flattening: [0]=voxel_features, [1..14]=weights, [15..70]=bn (g,b,m,v)*14,
  // [n_in-2]=voxel_coords, [n_in-1]=batch_size.
  static const int LC[14][5] = {  // cin, cout, kz, ky, kx (for ws sizing/prepack)
    { 4, 16, 3,3,3}, {16, 16, 3,3,3}, {16, 32, 3,3,3}, {32, 32, 3,3,3},
    {32, 32, 3,3,3}, {32, 64, 3,3,3}, {64, 64, 3,3,3}, {64, 64, 3,3,3},
    {64, 64, 3,3,3}, {64, 64, 3,3,3}, {64, 64, 3,3,3}, {16, 32, 3,3,3},
    {32, 64, 3,3,3}, {64,128, 3,1,1},
  };
  const size_t NF = (size_t)41*160*160, NH = (size_t)21*80*80;
  const size_t NQ = (size_t)11*40*40,  NE = (size_t)5*20*20, NO = (size_t)2*20*20;

  size_t off = 0;
  auto alloc = [&](size_t bytes) -> void* {
    off = (off + 255) & ~(size_t)255;
    void* ptr = (char*)d_ws + off;
    off += bytes;
    return ptr;
  };

  _Float16* in4    = (_Float16*)alloc(NF *  4 * 2);
  _Float16* fullA  = (_Float16*)alloc(NF * 16 * 2);
  _Float16* fullB  = (_Float16*)alloc(NF * 16 * 2);   // x1 (kept for branch)
  _Float16* full32 = (_Float16*)alloc(NF * 32 * 2);
  _Float16* full64 = (_Float16*)alloc(NF * 64 * 2);   // xp
  _Float16* halfA  = (_Float16*)alloc(NH * 32 * 2);
  _Float16* halfB  = (_Float16*)alloc(NH * 32 * 2);
  _Float16* qA     = (_Float16*)alloc(NQ * 64 * 2);
  _Float16* qB     = (_Float16*)alloc(NQ * 64 * 2);
  _Float16* eA     = (_Float16*)alloc(NE * 64 * 2);
  _Float16* eB     = (_Float16*)alloc(NE * 64 * 2);
  unsigned char* m1 = (unsigned char*)alloc(NF);
  unsigned char* m2 = (unsigned char*)alloc(NH);
  unsigned char* m3 = (unsigned char*)alloc(NQ);
  unsigned char* m4 = (unsigned char*)alloc(NE);
  unsigned char* mo = (unsigned char*)alloc(NO);
  _Float16* wp[14];
  for (int i = 0; i < 14; ++i) {
    int ntaps = LC[i][2] * LC[i][3] * LC[i][4];
    int kp = ((ntaps * LC[i][0] + 31) / 32) * 32;
    int ntiles = (LC[i][1] + 15) / 16;
    wp[i] = (_Float16*)alloc((size_t)kp * ntiles * 16 * 2);
  }
  size_t used = off;
  if (used > ws_size) used = ws_size;
  hipMemsetAsync(d_ws, 0, used, stream);
  hipMemsetAsync(d_out, 0, (size_t)out_size * sizeof(float), stream);

  // weight prepack into WMMA B-fragment layout (runs every call; tiny)
  for (int i = 0; i < 14; ++i) {
    int cin = LC[i][0], cout = LC[i][1];
    int ntaps = LC[i][2] * LC[i][3] * LC[i][4];
    int kp = ((ntaps * cin + 31) / 32) * 32;
    int ntiles = (cout + 15) / 16;
    int total = kp * ntiles * 16;
    prepack_wmma_b<<<(total + 255) / 256, 256, 0, stream>>>(
        (const float*)d_in[1 + i], wp[i], cin, cout, ntaps, kp);
  }

  const int N = in_sizes[0] / 4;
  const int* coords = (const int*)d_in[n_in - 2];
  scatter_voxels<<<(N + 255) / 256, 256, 0, stream>>>(
      (const float*)d_in[0], coords, in4, m1, N);

  auto dilate = [&](const unsigned char* in, unsigned char* out,
                    int iz,int iy,int ix, int oz,int oy,int ox,
                    int kz,int ky,int kx, int sz,int sy,int sx, int pz,int py,int px) {
    DilP p{in, out, iz,iy,ix, oz,oy,ox, kz,ky,kx, sz,sy,sx, pz,py,px};
    int nv = oz * oy * ox;
    mask_dilate<<<(nv + 255) / 256, 256, 0, stream>>>(p);
  };
  dilate(m1, m2, 41,160,160, 21,80,80, 3,3,3, 2,2,2, 1,1,1);
  dilate(m2, m3, 21,80,80,  11,40,40, 3,3,3, 2,2,2, 1,1,1);
  dilate(m3, m4, 11,40,40,  5,20,20,  3,3,3, 2,2,2, 0,1,1);
  dilate(m4, mo, 5,20,20,   2,20,20,  3,1,1, 2,1,1, 0,0,0);

  // conv launches: template args = CIN,COUT,KZ,KY,KX,SZ,SY,SX,PZ,PY,PX,
  //                                IZ,IY,IX,OZ,OY,OX,F32OUT
#define CONV(LI, IN, OUT, CINp,COUTp, KZp,KYp,KXp, SZp,SYp,SXp, PZp,PYp,PXp, \
             IZp,IYp,IXp, OZp,OYp,OXp, F32p, MASK)                            \
  conv_wmma_t<CINp,COUTp,KZp,KYp,KXp,SZp,SYp,SXp,PZp,PYp,PXp,                 \
              IZp,IYp,IXp,OZp,OYp,OXp,F32p>                                   \
      <<<dim3((OZp*OYp*OXp)/16, ((COUTp/16)+3)/4), 128, 0, stream>>>(         \
          IN, OUT, wp[LI], MASK,                                              \
          (const float*)d_in[15+4*(LI)+0], (const float*)d_in[15+4*(LI)+1],   \
          (const float*)d_in[15+4*(LI)+2], (const float*)d_in[15+4*(LI)+3])

  CONV( 0, in4,    fullA,   4, 16, 3,3,3, 1,1,1, 1,1,1, 41,160,160, 41,160,160, 0, m1);
  CONV( 1, fullA,  fullB,  16, 16, 3,3,3, 1,1,1, 1,1,1, 41,160,160, 41,160,160, 0, m1);
  CONV( 2, fullB,  halfA,  16, 32, 3,3,3, 2,2,2, 1,1,1, 41,160,160, 21, 80, 80, 0, m2);
  CONV( 3, halfA,  halfB,  32, 32, 3,3,3, 1,1,1, 1,1,1, 21, 80, 80, 21, 80, 80, 0, m2);
  CONV( 4, halfB,  halfA,  32, 32, 3,3,3, 1,1,1, 1,1,1, 21, 80, 80, 21, 80, 80, 0, m2);
  CONV( 5, halfA,  qA,     32, 64, 3,3,3, 2,2,2, 1,1,1, 21, 80, 80, 11, 40, 40, 0, m3);
  CONV( 6, qA,     qB,     64, 64, 3,3,3, 1,1,1, 1,1,1, 11, 40, 40, 11, 40, 40, 0, m3);
  CONV( 7, qB,     qA,     64, 64, 3,3,3, 1,1,1, 1,1,1, 11, 40, 40, 11, 40, 40, 0, m3);
  CONV( 8, qA,     eA,     64, 64, 3,3,3, 2,2,2, 0,1,1, 11, 40, 40,  5, 20, 20, 0, m4);
  CONV( 9, eA,     eB,     64, 64, 3,3,3, 1,1,1, 1,1,1,  5, 20, 20,  5, 20, 20, 0, m4);
  CONV(10, eB,     eA,     64, 64, 3,3,3, 1,1,1, 1,1,1,  5, 20, 20,  5, 20, 20, 0, m4);
  CONV(11, fullB,  full32, 16, 32, 3,3,3, 1,1,1, 1,1,1, 41,160,160, 41,160,160, 0, m1);
  CONV(12, full32, full64, 32, 64, 3,3,3, 1,1,1, 1,1,1, 41,160,160, 41,160,160, 0, m1);

  float* out_t = (float*)d_out;
  CONV(13, eA, out_t,      64,128, 3,1,1, 2,1,1, 0,0,0,  5, 20, 20,  2, 20, 20, 1, mo);
#undef CONV

  float* pf = out_t + NO * 128;
  float* pc = pf + (size_t)N * 64;
  gather_points<<<(N * 8 + 255) / 256, 256, 0, stream>>>(full64, coords, pf, pc, N);
}
